// Model_10402410791821
// MI455X (gfx1250) — matrix operand order repaired and checked by reading, not compiled, not run on hardware
//
#include <hip/hip_runtime.h>
#include <hip/hip_bf16.h>

typedef _Float16 half_t;
typedef __attribute__((ext_vector_type(16))) _Float16 v16h;
typedef __attribute__((ext_vector_type(8)))  float    v8f;

constexpr int Bn = 32, Qn = 32, Pn = 256, En = 512, Hn = 256;
constexpr int DDn = 512, G3n = 768, An_ = 75, ApadN = 80, XCn = 1024, G3P = 1536;

// ---- params in sorted-key order (jax dict flattening / _init_params insertion order) ----
enum {
  PE_B_WG = 0, PE_B_WH, PE_B_WI, PE_B_WK, PE_B_WQH, PE_B_WQI, PE_B_BH, PE_B_BI, PE_B_V,
  PE_F_WG, PE_F_WH, PE_F_WI, PE_F_WK, PE_F_WQH, PE_F_WQI, PE_F_BH, PE_F_BI, PE_F_V,
  PT_VQ, PT_WH, PT_WI, PT_BH, PT_BI, PT_P_WK, PT_P_WQ, PT_P_V, PT_Q_WK, PT_Q_WQ, PT_Q_V,
  SENC_P_B_WH, SENC_P_B_WI, SENC_P_B_BH, SENC_P_B_BI,
  SENC_P_F_WH, SENC_P_F_WI, SENC_P_F_BH, SENC_P_F_BI,
  SENC_Q_B_WH, SENC_Q_B_WI, SENC_Q_B_BH, SENC_Q_B_BI,
  SENC_Q_F_WH, SENC_Q_F_WI, SENC_Q_F_BH, SENC_Q_F_BI,
  SM_B_WG, SM_B_WH, SM_B_WI, SM_B_WK, SM_B_WQ, SM_B_BH, SM_B_BI, SM_B_V,
  SM_F_WG, SM_F_WH, SM_F_WI, SM_F_WK, SM_F_WQ, SM_F_BH, SM_F_BI, SM_F_V,
  NPARAM
};

struct WSLayout {
  half_t* senc_Wi[4]; half_t* senc_Wh[4];                 // [G3][E], [G3][H]
  half_t* at_Wk[4]; half_t* at_Wq[4]; half_t* at_Wqh[2];  // [80][512], [80][512], [80][256]
  half_t* at_Wg[4]; half_t* at_Wi[4]; half_t* at_Wh[4];   // [1024][1024], [768][1024], [768][256]
  half_t* ptq_Wk; half_t* ptp_Wk;                         // [512][512]
  half_t* q_enc; half_t* p_enc; half_t* vP; half_t* hP;   // [T][B][512] f16
  float* keyproj_pe; float* uproj_pe;                     // [2][Q][B][80], [2][P][B][80]
  float* keyproj_sm; float* uproj_sm;                     // [2][P][B][80]
  float* kproj_ptq; float* kproj_ptp;                     // [Q*B][512], [P*B][512]
  float* slab[4]; float* ptr_slab;
};

struct Ptrs {
  const float* in[4];
  const float* prm[NPARAM];
  WSLayout ws;
  float* out;
};

// ------------------------- WMMA helpers -------------------------
union V16U { v16h v; float4 f4[2]; };

// A fragment: 16x32 (MxK), 16-bit. lane<16: M=lane, K = k0+{0..7,16..23};
// lane>=16: M=lane-16, K = k0+{8..15,24..31}.
__device__ inline v16h load_fragA(const half_t* base, int ld, int m0, int k0) {
  int lane = threadIdx.x & 31;
  const half_t* p = base + (size_t)(m0 + (lane & 15)) * ld + (k0 + ((lane & 16) ? 8 : 0));
  V16U u;
  u.f4[0] = *reinterpret_cast<const float4*>(p);
  u.f4[1] = *reinterpret_cast<const float4*>(p + 16);
  return u.v;
}

// B fragment: 32x16 (KxN), weight stored row-major [N][K] (so B(k,n)=W[n][k]).
// lane<16: N=lane, K=k0+{0..7,16..23}; lane>=16: N=lane-16, K=k0+{8..15,24..31}.
__device__ inline v16h load_fragB(const half_t* base, int ld, int n0, int k0) {
  int lane = threadIdx.x & 31;
  const half_t* p = base + (size_t)(n0 + (lane & 15)) * ld + (k0 + ((lane & 16) ? 8 : 0));
  V16U u;
  u.f4[0] = *reinterpret_cast<const float4*>(p);
  u.f4[1] = *reinterpret_cast<const float4*>(p + 16);
  return u.v;
}

// D/C layout: element r of v8f, lane L: row = m0 + r + (L&16 ? 8 : 0), col = n0 + (L&15)
__device__ inline void store_tile(float* dst, int ld, int m0, int n0, v8f c) {
  int lane = threadIdx.x & 31;
  int hi = (lane & 16) ? 8 : 0;
  int n = n0 + (lane & 15);
#pragma unroll
  for (int r = 0; r < 8; ++r) dst[(size_t)(m0 + r + hi) * ld + n] = c[r];
}

__device__ inline v8f wmma_mm(const half_t* X, int ldx, const half_t* W, int ldw,
                              int m0, int n0, int K) {
  v8f acc = {0.f, 0.f, 0.f, 0.f, 0.f, 0.f, 0.f, 0.f};
  for (int k = 0; k < K; k += 32) {
    v16h a = load_fragA(X, ldx, m0, k);
    v16h b = load_fragB(W, ldw, n0, k);
    acc = __builtin_amdgcn_wmma_f32_16x16x32_f16(false, a, false, b, (short)0, acc,
                                                 false, false);
  }
  return acc;
}

__device__ inline float sigmoidf_(float x) { return 1.f / (1.f + __expf(-x)); }

#define FB() do { __threadfence_block(); __syncthreads(); } while (0)

// wave-per-row softmax over rows of m[Bn][KT] (Bn == 32 == #waves)
__device__ inline void softmax_rows(float* m, int KT) {
  int wave = threadIdx.x >> 5, lane = threadIdx.x & 31;
  float* row = m + (size_t)wave * KT;
  float mx = -1e30f;
  for (int k = lane; k < KT; k += 32) mx = fmaxf(mx, row[k]);
  for (int o = 16; o; o >>= 1) mx = fmaxf(mx, __shfl_xor(mx, o, 32));
  float sum = 0.f;
  for (int k = lane; k < KT; k += 32) { float e = __expf(row[k] - mx); row[k] = e; sum += e; }
  for (int o = 16; o; o >>= 1) sum += __shfl_xor(sum, o, 32);
  float inv = 1.f / sum;
  for (int k = lane; k < KT; k += 32) row[k] *= inv;
}

// ------------------------- weight f32 -> f16 (padded) -------------------------
__global__ void k_cvt(const float* src, half_t* dst, int rs, int cs, int rd, int cd) {
  int i = blockIdx.x * blockDim.x + threadIdx.x;
  if (i >= rd * cd) return;
  int r = i / cd, c = i % cd;
  float v = (src && r < rs && c < cs) ? src[(size_t)r * cs + c] : 0.f;
  dst[i] = (half_t)v;
}

// ------------------------- generic O[M][N] = X[M][K] @ W[N][K]^T -------------------------
__global__ __launch_bounds__(256) void k_matmul(const half_t* X, const half_t* W, float* O,
                                                int M, int N, int K) {
  int wave = threadIdx.x >> 5;
  int tile = blockIdx.x * 8 + wave;
  int nt = N / 16;
  if (tile >= (M / 16) * nt) return;  // wave-uniform; EXEC stays all-ones in live waves
  int m0 = (tile / nt) * 16, n0 = (tile % nt) * 16;
  v8f acc = wmma_mm(X, K, W, K, m0, n0, K);
  store_tile(O, N, m0, n0, acc);
}

// ------------------------- bidirectional sentence encoders -------------------------
__device__ inline const float* enc_bi_p(const Ptrs& p, int e) {
  switch (e) { case 0: return p.prm[SENC_Q_F_BI]; case 1: return p.prm[SENC_Q_B_BI];
               case 2: return p.prm[SENC_P_F_BI]; default: return p.prm[SENC_P_B_BI]; }
}
__device__ inline const float* enc_bh_p(const Ptrs& p, int e) {
  switch (e) { case 0: return p.prm[SENC_Q_F_BH]; case 1: return p.prm[SENC_Q_B_BH];
               case 2: return p.prm[SENC_P_F_BH]; default: return p.prm[SENC_P_B_BH]; }
}

__global__ __launch_bounds__(1024) void k_encoder(Ptrs p) {
  int e = blockIdx.x;                 // 0=q_f 1=q_b 2=p_f 3=p_b
  int T = (e < 2) ? Qn : Pn;
  int dirb = e & 1;
  const float* X = (e < 2) ? p.in[0] : p.in[1];          // [T][B][E]
  half_t* out = (e < 2) ? p.ws.q_enc : p.ws.p_enc;        // [T][B][DD]
  const half_t* Wi = p.ws.senc_Wi[e];
  const half_t* Wh = p.ws.senc_Wh[e];
  const float* bi = enc_bi_p(p, e);
  const float* bh = enc_bh_p(p, e);

  char* slab = (char*)p.ws.slab[e];
  half_t* x_h = (half_t*)slab;                              // 32*512 h
  half_t* h_h = (half_t*)(slab + Bn * En * 2);              // 32*256 h
  float*  h_f = (float*)(slab + Bn * En * 2 + Bn * Hn * 2); // 32*256 f
  float*  gi  = h_f + Bn * Hn;                              // 32*768 f
  float*  gh  = gi + Bn * G3n;                              // 32*768 f

  int tid = threadIdx.x, wave = tid >> 5;
  for (int i = tid; i < Bn * Hn; i += 1024) { h_f[i] = 0.f; h_h[i] = (half_t)0.f; }
  FB();

  for (int t = 0; t < T; ++t) {
    int ts = dirb ? (T - 1 - t) : t;
    const float* xt = X + (size_t)ts * Bn * En;
    for (int i = tid; i < Bn * En; i += 1024) x_h[i] = (half_t)xt[i];
    FB();
    // gi = x @ Wi^T (96 tiles, K=512); gh = h @ Wh^T (96 tiles, K=256)
    for (int tile = wave; tile < 192; tile += 32) {
      bool isGi = tile < 96;
      int tt = isGi ? tile : tile - 96;
      int m0 = (tt & 1) * 16, n0 = (tt >> 1) * 16;
      v8f acc = isGi ? wmma_mm(x_h, En, Wi, En, m0, n0, En)
                     : wmma_mm(h_h, Hn, Wh, Hn, m0, n0, Hn);
      store_tile(isGi ? gi : gh, G3n, m0, n0, acc);
    }
    FB();
    for (int i = tid; i < Bn * Hn; i += 1024) {
      int b = i >> 8, j = i & 255;
      float ir = gi[b * G3n + j] + bi[j];
      float iz = gi[b * G3n + Hn + j] + bi[Hn + j];
      float in_ = gi[b * G3n + 2 * Hn + j] + bi[2 * Hn + j];
      float hr = gh[b * G3n + j] + bh[j];
      float hz = gh[b * G3n + Hn + j] + bh[Hn + j];
      float hn = gh[b * G3n + 2 * Hn + j] + bh[2 * Hn + j];
      float r = sigmoidf_(ir + hr);
      float z = sigmoidf_(iz + hz);
      float n = tanhf(in_ + r * hn);
      float h = (1.f - z) * n + z * h_f[i];
      h_f[i] = h; h_h[i] = (half_t)h;
      out[(size_t)ts * Bn * DDn + (size_t)b * DDn + dirb * Hn + j] = (half_t)h;
    }
    FB();
  }
}

// ------------------------- gated attention GRUs (pe: stage 0, sm: stage 1) -------------------------
__device__ inline const float* at_v_p(const Ptrs& p, int s) {
  switch (s) { case 0: return p.prm[PE_F_V]; case 1: return p.prm[PE_B_V];
               case 2: return p.prm[SM_F_V]; default: return p.prm[SM_B_V]; }
}
__device__ inline const float* at_bi_p(const Ptrs& p, int s) {
  switch (s) { case 0: return p.prm[PE_F_BI]; case 1: return p.prm[PE_B_BI];
               case 2: return p.prm[SM_F_BI]; default: return p.prm[SM_B_BI]; }
}
__device__ inline const float* at_bh_p(const Ptrs& p, int s) {
  switch (s) { case 0: return p.prm[PE_F_BH]; case 1: return p.prm[PE_B_BH];
               case 2: return p.prm[SM_F_BH]; default: return p.prm[SM_B_BH]; }
}

__global__ __launch_bounds__(1024) void k_attn(Ptrs p, int stage) {
  int d = blockIdx.x;                  // 0=f, 1=b
  int slot = stage * 2 + d;
  int KT = stage ? Pn : Qn;
  const half_t* keys = stage ? p.ws.vP : p.ws.q_enc;   // [KT][B][DD]
  const half_t* U    = stage ? p.ws.vP : p.ws.p_enc;   // [P][B][DD]
  half_t* out        = stage ? p.ws.hP : p.ws.vP;
  const float* kp = (stage ? p.ws.keyproj_sm : p.ws.keyproj_pe) + (size_t)d * KT * Bn * ApadN;
  const float* up = (stage ? p.ws.uproj_sm   : p.ws.uproj_pe)   + (size_t)d * Pn * Bn * ApadN;
  const float* kmask = p.in[stage ? 3 : 2];            // [B][KT]
  const half_t* Wg = p.ws.at_Wg[slot];
  const half_t* Wi = p.ws.at_Wi[slot];
  const half_t* Wh = p.ws.at_Wh[slot];
  const half_t* Wqh = (stage == 0) ? p.ws.at_Wqh[d] : nullptr;
  const float* v  = at_v_p(p, slot);
  const float* bi = at_bi_p(p, slot);
  const float* bh = at_bh_p(p, slot);

  char* slab = (char*)p.ws.slab[slot];
  half_t* xc_h = (half_t*)slab;                     // 32*1024 h  (64KB)
  half_t* h_h  = (half_t*)(slab + (64 << 10));      // 32*256 h
  float*  h_f  = (float*)(slab + (80 << 10));       // 32*256 f
  float*  qph  = (float*)(slab + (112 << 10));      // 32*80  f
  float*  probs= (float*)(slab + (128 << 10));      // 32*256 f
  float*  gate = (float*)(slab + (160 << 10));      // 32*1024 f
  float*  gi   = (float*)(slab + (288 << 10));      // 32*768 f
  float*  gh   = (float*)(slab + (384 << 10));      // 32*768 f

  int tid = threadIdx.x, wave = tid >> 5;
  for (int i = tid; i < Bn * Hn; i += 1024) { h_f[i] = 0.f; h_h[i] = (half_t)0.f; }
  for (int i = tid; i < Bn * ApadN; i += 1024) qph[i] = 0.f;
  FB();

  for (int t = 0; t < Pn; ++t) {
    int ts = d ? (Pn - 1 - t) : t;
    // phase 0: qp_h = h @ Wqh^T (pe only)
    if (stage == 0) {
      for (int tile = wave; tile < 10; tile += 32) {
        int m0 = (tile & 1) * 16, n0 = (tile >> 1) * 16;
        v8f acc = wmma_mm(h_h, Hn, Wqh, Hn, m0, n0, Hn);
        store_tile(qph, ApadN, m0, n0, acc);
      }
      FB();
    }
    // phase 1: scores
    const float* upt = up + (size_t)ts * Bn * ApadN;
    for (int i = tid; i < Bn * KT; i += 1024) {
      int b = i / KT, tk = i % KT;
      const float* kpr = kp + ((size_t)tk * Bn + b) * ApadN;
      const float* qpr = qph + b * ApadN;
      const float* upr = upt + b * ApadN;
      float s = 0.f;
      for (int a = 0; a < An_; ++a) s += v[a] * tanhf(kpr[a] + qpr[a] + upr[a]);
      probs[b * KT + tk] = (kmask[b * KT + tk] > 0.f) ? s : -1e9f;
    }
    FB();
    softmax_rows(probs, KT);
    FB();
    // phase 3: xc = [u, c]
    const half_t* ut = U + (size_t)ts * Bn * DDn;
    for (int i = tid; i < Bn * DDn; i += 1024)
      xc_h[(i >> 9) * XCn + (i & 511)] = ut[i];
    for (int i = tid; i < Bn * DDn; i += 1024) {
      int b = i >> 9, dd = i & 511;
      float c = 0.f;
      for (int tk = 0; tk < KT; ++tk)
        c += probs[b * KT + tk] * (float)keys[((size_t)tk * Bn + b) * DDn + dd];
      xc_h[b * XCn + DDn + dd] = (half_t)c;
    }
    FB();
    // phase 4: gate = xc @ Wg^T  (2x64 tiles, K=1024)
    for (int tile = wave; tile < 128; tile += 32) {
      int m0 = (tile & 1) * 16, n0 = (tile >> 1) * 16;
      v8f acc = wmma_mm(xc_h, XCn, Wg, XCn, m0, n0, XCn);
      store_tile(gate, XCn, m0, n0, acc);
    }
    FB();
    // phase 5: xc *= sigmoid(gate)
    for (int i = tid; i < Bn * XCn; i += 1024)
      xc_h[i] = (half_t)((float)xc_h[i] * sigmoidf_(gate[i]));
    FB();
    // phase 6: gi = xc @ Wi^T (K=1024); gh = h @ Wh^T (K=256)
    for (int tile = wave; tile < 192; tile += 32) {
      bool g1 = tile < 96;
      int tt = g1 ? tile : tile - 96;
      int m0 = (tt & 1) * 16, n0 = (tt >> 1) * 16;
      v8f acc = g1 ? wmma_mm(xc_h, XCn, Wi, XCn, m0, n0, XCn)
                   : wmma_mm(h_h, Hn, Wh, Hn, m0, n0, Hn);
      store_tile(g1 ? gi : gh, G3n, m0, n0, acc);
    }
    FB();
    // phase 7: GRU update
    for (int i = tid; i < Bn * Hn; i += 1024) {
      int b = i >> 8, j = i & 255;
      float ir = gi[b * G3n + j] + bi[j];
      float iz = gi[b * G3n + Hn + j] + bi[Hn + j];
      float in_ = gi[b * G3n + 2 * Hn + j] + bi[2 * Hn + j];
      float hr = gh[b * G3n + j] + bh[j];
      float hz = gh[b * G3n + Hn + j] + bh[Hn + j];
      float hn = gh[b * G3n + 2 * Hn + j] + bh[2 * Hn + j];
      float r = sigmoidf_(ir + hr);
      float z = sigmoidf_(iz + hz);
      float n = tanhf(in_ + r * hn);
      float h = (1.f - z) * n + z * h_f[i];
      h_f[i] = h; h_h[i] = (half_t)h;
      out[(size_t)ts * Bn * DDn + (size_t)b * DDn + d * Hn + j] = (half_t)h;
    }
    FB();
  }
}

// ------------------------- pointer network (single block) -------------------------
__global__ __launch_bounds__(1024) void k_ptr(Ptrs p) {
  float* sl = p.ws.ptr_slab;
  float* wqvq = sl;                     // 512
  float* rq   = wqvq + DDn;             // 32*512
  float* qb   = rq + Bn * DDn;          // 32*512
  float* probs= qb + Bn * DDn;          // 32*256
  float* c1   = probs + Bn * Pn;        // 32*512
  float* gi   = c1 + Bn * DDn;          // 32*1536
  float* gh   = gi + Bn * G3P;          // 32*1536
  float* h1   = gh + Bn * G3P;          // 32*512

  const float* Vq  = p.prm[PT_VQ];
  const float* qWq = p.prm[PT_Q_WQ]; const float* qv = p.prm[PT_Q_V];
  const float* pWq = p.prm[PT_P_WQ]; const float* pv = p.prm[PT_P_V];
  const float* Wi  = p.prm[PT_WI];   const float* Wh = p.prm[PT_WH];
  const float* bi  = p.prm[PT_BI];   const float* bh = p.prm[PT_BH];
  const float* qmask = p.in[2]; const float* pmask = p.in[3];
  const float* kq = p.ws.kproj_ptq; const float* kp = p.ws.kproj_ptp;
  const half_t* qenc = p.ws.q_enc; const half_t* hP = p.ws.hP;
  int tid = threadIdx.x;

  // 1: wqvq = Wq @ Vq
  for (int a = tid; a < DDn; a += 1024) {
    float s = 0.f;
    for (int dd = 0; dd < DDn; ++dd) s += qWq[(size_t)a * DDn + dd] * Vq[dd];
    wqvq[a] = s;
  }
  FB();
  // 2: question pooling scores
  for (int i = tid; i < Bn * Qn; i += 1024) {
    int b = i / Qn, tk = i % Qn;
    const float* kr = kq + ((size_t)tk * Bn + b) * DDn;
    float s = 0.f;
    for (int a = 0; a < DDn; ++a) s += qv[a] * tanhf(kr[a] + wqvq[a]);
    probs[b * Qn + tk] = (qmask[b * Qn + tk] > 0.f) ? s : -1e9f;
  }
  FB();
  softmax_rows(probs, Qn);
  FB();
  // 3: r_q
  for (int i = tid; i < Bn * DDn; i += 1024) {
    int b = i / DDn, dd = i % DDn;
    float s = 0.f;
    for (int tk = 0; tk < Qn; ++tk)
      s += probs[b * Qn + tk] * (float)qenc[((size_t)tk * Bn + b) * DDn + dd];
    rq[i] = s;
  }
  FB();
  // 4: qb = r_q @ pWq^T
  for (int i = tid; i < Bn * DDn; i += 1024) {
    int b = i / DDn, a = i % DDn;
    float s = 0.f;
    for (int dd = 0; dd < DDn; ++dd) s += pWq[(size_t)a * DDn + dd] * rq[b * DDn + dd];
    qb[i] = s;
  }
  FB();
  // 5: begin logits
  for (int i = tid; i < Bn * Pn; i += 1024) {
    int b = i / Pn, t = i % Pn;
    const float* kr = kp + ((size_t)t * Bn + b) * DDn;
    const float* qr = qb + b * DDn;
    float s = 0.f;
    for (int a = 0; a < DDn; ++a) s += pv[a] * tanhf(kr[a] + qr[a]);
    s = (pmask[b * Pn + t] > 0.f) ? s : -1e9f;
    p.out[b * Pn + t] = s;
    probs[b * Pn + t] = s;
  }
  FB();
  softmax_rows(probs, Pn);
  FB();
  // 6: c1
  for (int i = tid; i < Bn * DDn; i += 1024) {
    int b = i / DDn, dd = i % DDn;
    float s = 0.f;
    for (int t = 0; t < Pn; ++t)
      s += probs[b * Pn + t] * (float)hP[((size_t)t * Bn + b) * DDn + dd];
    c1[i] = s;
  }
  FB();
  // 7: GRU(c1, r_q)
  for (int i = tid; i < Bn * G3P; i += 1024) {
    int b = i / G3P, g = i % G3P;
    float s1 = bi[g], s2 = bh[g];
    for (int dd = 0; dd < DDn; ++dd) {
      s1 += Wi[(size_t)g * DDn + dd] * c1[b * DDn + dd];
      s2 += Wh[(size_t)g * DDn + dd] * rq[b * DDn + dd];
    }
    gi[i] = s1; gh[i] = s2;
  }
  FB();
  for (int i = tid; i < Bn * DDn; i += 1024) {
    int b = i / DDn, j = i % DDn;
    float ir = gi[b * G3P + j],        iz = gi[b * G3P + DDn + j],  in_ = gi[b * G3P + 2 * DDn + j];
    float hr = gh[b * G3P + j],        hz = gh[b * G3P + DDn + j],  hn  = gh[b * G3P + 2 * DDn + j];
    float r = sigmoidf_(ir + hr);
    float z = sigmoidf_(iz + hz);
    float n = tanhf(in_ + r * hn);
    h1[i] = (1.f - z) * n + z * rq[i];
  }
  FB();
  // 8: qe = h1 @ pWq^T (reuse qb)
  for (int i = tid; i < Bn * DDn; i += 1024) {
    int b = i / DDn, a = i % DDn;
    float s = 0.f;
    for (int dd = 0; dd < DDn; ++dd) s += pWq[(size_t)a * DDn + dd] * h1[b * DDn + dd];
    qb[i] = s;
  }
  FB();
  // 9: end logits
  for (int i = tid; i < Bn * Pn; i += 1024) {
    int b = i / Pn, t = i % Pn;
    const float* kr = kp + ((size_t)t * Bn + b) * DDn;
    const float* qr = qb + b * DDn;
    float s = 0.f;
    for (int a = 0; a < DDn; ++a) s += pv[a] * tanhf(kr[a] + qr[a]);
    p.out[Bn * Pn + b * Pn + t] = (pmask[b * Pn + t] > 0.f) ? s : -1e9f;
  }
}

// ------------------------- host -------------------------
extern "C" void kernel_launch(void* const* d_in, const int* in_sizes, int n_in,
                              void* d_out, int out_size, void* d_ws, size_t ws_size,
                              hipStream_t stream) {
  if (n_in < 4 + NPARAM) return;

  Ptrs pk;
  for (int i = 0; i < 4; ++i) pk.in[i] = (const float*)d_in[i];
  for (int i = 0; i < NPARAM; ++i) pk.prm[i] = (const float*)d_in[4 + i];
  pk.out = (float*)d_out;

  // workspace bump allocator (deterministic layout every call)
  char* base = (char*)d_ws;
  size_t off = 0;
  auto alloc = [&](size_t bytes) -> char* {
    char* r = base + off;
    off = (off + bytes + 255) & ~(size_t)255;
    return r;
  };
  WSLayout& w = pk.ws;
  for (int i = 0; i < 4; ++i) {
    w.senc_Wi[i] = (half_t*)alloc((size_t)G3n * En * 2);
    w.senc_Wh[i] = (half_t*)alloc((size_t)G3n * Hn * 2);
  }
  for (int i = 0; i < 4; ++i) {
    w.at_Wk[i] = (half_t*)alloc((size_t)ApadN * DDn * 2);
    w.at_Wq[i] = (half_t*)alloc((size_t)ApadN * DDn * 2);
    w.at_Wg[i] = (half_t*)alloc((size_t)XCn * XCn * 2);
    w.at_Wi[i] = (half_t*)alloc((size_t)G3n * XCn * 2);
    w.at_Wh[i] = (half_t*)alloc((size_t)G3n * Hn * 2);
  }
  for (int i = 0; i < 2; ++i) w.at_Wqh[i] = (half_t*)alloc((size_t)ApadN * Hn * 2);
  w.ptq_Wk = (half_t*)alloc((size_t)DDn * DDn * 2);
  w.ptp_Wk = (half_t*)alloc((size_t)DDn * DDn * 2);
  w.q_enc = (half_t*)alloc((size_t)Qn * Bn * DDn * 2);
  w.p_enc = (half_t*)alloc((size_t)Pn * Bn * DDn * 2);
  w.vP = (half_t*)alloc((size_t)Pn * Bn * DDn * 2);
  w.hP = (half_t*)alloc((size_t)Pn * Bn * DDn * 2);
  w.keyproj_pe = (float*)alloc((size_t)2 * Qn * Bn * ApadN * 4);
  w.uproj_pe = (float*)alloc((size_t)2 * Pn * Bn * ApadN * 4);
  w.keyproj_sm = (float*)alloc((size_t)2 * Pn * Bn * ApadN * 4);
  w.uproj_sm = (float*)alloc((size_t)2 * Pn * Bn * ApadN * 4);
  w.kproj_ptq = (float*)alloc((size_t)Qn * Bn * DDn * 4);
  w.kproj_ptp = (float*)alloc((size_t)Pn * Bn * DDn * 4);
  for (int i = 0; i < 4; ++i) w.slab[i] = (float*)alloc(512 * 1024);
  w.ptr_slab = (float*)alloc(1024 * 1024);

  // ---- weight conversions ----
  struct Cvt { const float* src; half_t* dst; int rs, cs, rd, cd; };
  Cvt cv[40]; int nc = 0;
  auto add = [&](int pi, half_t* dst, int rs, int cs, int rd, int cd) {
    cv[nc++] = {pk.prm[pi], dst, rs, cs, rd, cd};
  };
  const int encWi[4] = {SENC_Q_F_WI, SENC_Q_B_WI, SENC_P_F_WI, SENC_P_B_WI};
  const int encWh[4] = {SENC_Q_F_WH, SENC_Q_B_WH, SENC_P_F_WH, SENC_P_B_WH};
  for (int i = 0; i < 4; ++i) {
    add(encWi[i], w.senc_Wi[i], G3n, En, G3n, En);
    add(encWh[i], w.senc_Wh[i], G3n, Hn, G3n, Hn);
  }
  const int atWk[4] = {PE_F_WK, PE_B_WK, SM_F_WK, SM_B_WK};
  const int atWq[4] = {PE_F_WQI, PE_B_WQI, SM_F_WQ, SM_B_WQ};
  const int atWg[4] = {PE_F_WG, PE_B_WG, SM_F_WG, SM_B_WG};
  const int atWi[4] = {PE_F_WI, PE_B_WI, SM_F_WI, SM_B_WI};
  const int atWh[4] = {PE_F_WH, PE_B_WH, SM_F_WH, SM_B_WH};
  for (int i = 0; i < 4; ++i) {
    add(atWk[i], w.at_Wk[i], An_, DDn, ApadN, DDn);
    add(atWq[i], w.at_Wq[i], An_, DDn, ApadN, DDn);
    add(atWg[i], w.at_Wg[i], XCn, XCn, XCn, XCn);
    add(atWi[i], w.at_Wi[i], G3n, XCn, G3n, XCn);
    add(atWh[i], w.at_Wh[i], G3n, Hn, G3n, Hn);
  }
  add(PE_F_WQH, w.at_Wqh[0], An_, Hn, ApadN, Hn);
  add(PE_B_WQH, w.at_Wqh[1], An_, Hn, ApadN, Hn);
  add(PT_Q_WK, w.ptq_Wk, DDn, DDn, DDn, DDn);
  add(PT_P_WK, w.ptp_Wk, DDn, DDn, DDn, DDn);
  for (int i = 0; i < nc; ++i) {
    int n = cv[i].rd * cv[i].cd;
    k_cvt<<<(n + 255) / 256, 256, 0, stream>>>(cv[i].src, cv[i].dst,
                                               cv[i].rs, cv[i].cs, cv[i].rd, cv[i].cd);
  }

  auto mm = [&](const half_t* X, const half_t* Wm, float* O, int M, int N, int K) {
    int tiles = (M / 16) * (N / 16);
    k_matmul<<<(tiles + 7) / 8, 256, 0, stream>>>(X, Wm, O, M, N, K);
  };

  // ---- pipeline ----
  k_encoder<<<4, 1024, 0, stream>>>(pk);

  mm(w.q_enc, w.at_Wk[0], w.keyproj_pe, Qn * Bn, ApadN, DDn);
  mm(w.q_enc, w.at_Wk[1], w.keyproj_pe + (size_t)Qn * Bn * ApadN, Qn * Bn, ApadN, DDn);
  mm(w.p_enc, w.at_Wq[0], w.uproj_pe, Pn * Bn, ApadN, DDn);
  mm(w.p_enc, w.at_Wq[1], w.uproj_pe + (size_t)Pn * Bn * ApadN, Pn * Bn, ApadN, DDn);
  k_attn<<<2, 1024, 0, stream>>>(pk, 0);

  mm(w.vP, w.at_Wk[2], w.keyproj_sm, Pn * Bn, ApadN, DDn);
  mm(w.vP, w.at_Wk[3], w.keyproj_sm + (size_t)Pn * Bn * ApadN, Pn * Bn, ApadN, DDn);
  mm(w.vP, w.at_Wq[2], w.uproj_sm, Pn * Bn, ApadN, DDn);
  mm(w.vP, w.at_Wq[3], w.uproj_sm + (size_t)Pn * Bn * ApadN, Pn * Bn, ApadN, DDn);
  k_attn<<<2, 1024, 0, stream>>>(pk, 1);

  mm(w.q_enc, w.ptq_Wk, w.kproj_ptq, Qn * Bn, DDn, DDn);
  mm(w.hP, w.ptp_Wk, w.kproj_ptp, Pn * Bn, DDn, DDn);
  k_ptr<<<1, 1024, 0, stream>>>(pk);

  (void)in_sizes; (void)out_size; (void)ws_size;
}
